// SeqAtt_34660386079012
// MI455X (gfx1250) — compile-verified, matrix operand order checked by reading, not run on hardware
//
#include <hip/hip_runtime.h>
#include <hip/hip_bf16.h>

// ---------------------------------------------------------------------------
// Fused LN + QKVG proj + multi-head attention + gate + output proj (MI455X)
// Matmuls: v_wmma_f32_16x16x32_f16 (wave32, fp32 accumulate).
// Tile staging: Tensor Data Mover (tensor_load_to_lds) + TENSORcnt waits.
// ---------------------------------------------------------------------------

typedef __attribute__((ext_vector_type(16))) _Float16 v16h;
typedef __attribute__((ext_vector_type(8)))  _Float16 v8h;
typedef __attribute__((ext_vector_type(8)))  float    v8f;
typedef __attribute__((ext_vector_type(4)))  float    v4f;
typedef __attribute__((ext_vector_type(4)))  unsigned u32x4;
typedef __attribute__((ext_vector_type(8)))  unsigned u32x8;

#define BATCH   2
#define LSEQ    2048
#define DMODEL  1024
#define NHEAD   16
#define HDIM    64
#define SROWS   (BATCH * LSEQ)     // 4096
#define N4      (4 * DMODEL)       // 4096  (Q|K|V|G concatenated columns)

__device__ __forceinline__ v8f wmma_f16(v16h a, v16h b, v8f c) {
    return __builtin_amdgcn_wmma_f32_16x16x32_f16(false, a, false, b,
                                                  (short)0, c, false, false);
}

// Build a 16-half A/B fragment from two 16-byte (8-half) chunks.
__device__ __forceinline__ v16h ld_frag(const _Float16* p0, const _Float16* p1) {
    v8h a = *reinterpret_cast<const v8h*>(p0);
    v8h b = *reinterpret_cast<const v8h*>(p1);
    return __builtin_shufflevector(a, b, 0, 1, 2, 3, 4, 5, 6, 7,
                                         8, 9, 10, 11, 12, 13, 14, 15);
}

// ---------------------------------------------------------------------------
// TDM: DMA a 2D f16 tile (tile_d0 elems/row, tile_d1 rows, row stride in
// elems) from global memory into LDS (contiguous, row-major).
// D# per CDNA5 ISA ch.8: group0 = {count/type/addr}, group1 = {dims/strides}.
// ---------------------------------------------------------------------------
__device__ __forceinline__ void tdm_load_tile_f16(const _Float16* gptr,
                                                  const _Float16* ldsptr,
                                                  unsigned tile_d0,
                                                  unsigned tile_d1,
                                                  unsigned stride_elems) {
    unsigned long long ga = (unsigned long long)(size_t)gptr;
    unsigned lds_addr = (unsigned)(size_t)ldsptr;      // LDS aperture low bits
    u32x4 g0;
    g0[0] = 1u;                                        // count=1, user D#
    g0[1] = lds_addr;                                  // lds_addr (bytes)
    g0[2] = (unsigned)ga;                              // global_addr[31:0]
    g0[3] = ((unsigned)(ga >> 32) & 0x01FFFFFFu)       // global_addr[56:32]
            | 0x80000000u;                             // type = 2 ("image")
    u32x8 g1;
    g1[0] = 0x00010000u;                               // data_size=1 (2 bytes)
    g1[1] = (tile_d0 & 0xFFFFu) << 16;                 // tensor_dim0[15:0]
    g1[2] = (tile_d0 >> 16) | ((tile_d1 & 0xFFFFu) << 16); // tdim0 hi | tdim1 lo
    g1[3] = (tile_d1 >> 16) | (tile_d0 << 16);         // tdim1 hi | tile_dim0
    g1[4] = tile_d1 & 0xFFFFu;                         // tile_dim1 (tile_dim2=0)
    g1[5] = stride_elems;                              // tensor_dim0_stride lo
    g1[6] = 0u;
    g1[7] = 0u;
    asm volatile("tensor_load_to_lds %0, %1" :: "s"(g0), "s"(g1) : "memory");
}

__device__ __forceinline__ void tdm_wait0() {
    asm volatile("s_wait_tensorcnt 0x0" ::: "memory");
}

// ---------------------------------------------------------------------------
// Kernel 1: weight conversion/packing to f16, TRANSPOSED (N x K) so that
// GEMM B tiles are natural row-major 2D tiles for the TDM.
// WallT (4096 x 1024) = [Wq|Wk|Wv|Wg]^T ; WohT (1024 x 1024) = Wo^T
// ---------------------------------------------------------------------------
__global__ void prep_weights(const float* __restrict__ Wq,
                             const float* __restrict__ Wk,
                             const float* __restrict__ Wv,
                             const float* __restrict__ Wg,
                             const float* __restrict__ Wo,
                             _Float16* __restrict__ WallT,
                             _Float16* __restrict__ WohT) {
    const int total = DMODEL * N4 + DMODEL * DMODEL;
    int i = blockIdx.x * blockDim.x + threadIdx.x;
    if (i >= total) return;
    if (i < DMODEL * N4) {
        int k   = i >> 12;          // row in D (source row, contiguous reads)
        int c   = i & (N4 - 1);     // packed output column
        int sel = c >> 10;
        int col = c & (DMODEL - 1);
        const float* src = (sel == 0) ? Wq : (sel == 1) ? Wk : (sel == 2) ? Wv : Wg;
        WallT[(size_t)c * DMODEL + k] = (_Float16)src[k * DMODEL + col];
    } else {
        int j = i - DMODEL * N4;
        int k = j >> 10;
        int n = j & (DMODEL - 1);
        WohT[(size_t)n * DMODEL + k] = (_Float16)Wo[j];
    }
}

// ---------------------------------------------------------------------------
// Kernel 2: LayerNorm over D=1024, emit f16 x-hat. One block per row.
// ---------------------------------------------------------------------------
__global__ __launch_bounds__(256) void layernorm_f16(const float* __restrict__ seq,
                                                     const float* __restrict__ gamma,
                                                     const float* __restrict__ beta,
                                                     _Float16* __restrict__ Xh) {
    __shared__ float red[256];
    const int row = blockIdx.x;
    const int tid = threadIdx.x;
    const float* p = seq + (size_t)row * DMODEL + tid * 4;
    v4f v = *reinterpret_cast<const v4f*>(p);

    red[tid] = v.x + v.y + v.z + v.w;
    __syncthreads();
    for (int off = 128; off > 0; off >>= 1) {
        if (tid < off) red[tid] += red[tid + off];
        __syncthreads();
    }
    const float mu = red[0] * (1.0f / DMODEL);
    __syncthreads();

    float d0 = v.x - mu, d1 = v.y - mu, d2 = v.z - mu, d3 = v.w - mu;
    red[tid] = d0 * d0 + d1 * d1 + d2 * d2 + d3 * d3;
    __syncthreads();
    for (int off = 128; off > 0; off >>= 1) {
        if (tid < off) red[tid] += red[tid + off];
        __syncthreads();
    }
    const float inv = rsqrtf(red[0] * (1.0f / DMODEL) + 1e-5f);

    v4f g = *reinterpret_cast<const v4f*>(gamma + tid * 4);
    v4f b = *reinterpret_cast<const v4f*>(beta + tid * 4);
    _Float16* op = Xh + (size_t)row * DMODEL + tid * 4;
    op[0] = (_Float16)(d0 * inv * g.x + b.x);
    op[1] = (_Float16)(d1 * inv * g.y + b.y);
    op[2] = (_Float16)(d2 * inv * g.z + b.z);
    op[3] = (_Float16)(d3 * inv * g.w + b.w);
}

// ---------------------------------------------------------------------------
// Kernel 3/5: WMMA GEMM  C(MxN) = A(MxK) @ B(KxN), B given TRANSPOSED (NxK).
// Block tile 128(M) x 64(N); 8 waves, each wave: one 16-row strip x 4 n-tiles.
// B tile (64 rows x 32 k-elems) staged into LDS by the Tensor Data Mover;
// fragments are then two ds_load_b128 each.
// mode 0: split epilogue -> Q/K/V f16 + gate=sigmoid(.+bg) f32   (N = 4096)
// mode 1: Y = . + bo -> f32 output                                (N = 1024)
// ---------------------------------------------------------------------------
__global__ __launch_bounds__(256) void wmma_gemm(const _Float16* __restrict__ A,
                                                 const _Float16* __restrict__ BwT,
                                                 int N, int Kd, int mode,
                                                 _Float16* __restrict__ Qo,
                                                 _Float16* __restrict__ Ko,
                                                 _Float16* __restrict__ Vo,
                                                 float* __restrict__ gateo,
                                                 const float* __restrict__ bg,
                                                 float* __restrict__ Yo,
                                                 const float* __restrict__ bo) {
    __shared__ __align__(16) _Float16 Bs[64 * 32];   // Bs[n][k], row-major

    const int tid  = threadIdx.x;
    const int lane = tid & 31;
    const int wave = tid >> 5;
    const int lo   = lane & 15;
    const int hi   = lane >> 4;

    const int nbase = blockIdx.x * 64;
    const int mbase = blockIdx.y * 128 + wave * 16;
    const _Float16* arow = A + (size_t)(mbase + lo) * Kd;

    v8f acc[4] = {v8f{}, v8f{}, v8f{}, v8f{}};

    for (int kb = 0; kb < Kd; kb += 32) {
        // TDM: stage 64(n) x 32(k) tile of B^T into LDS (one DMA per block)
        if (wave == 0) {
            tdm_load_tile_f16(BwT + (size_t)nbase * Kd + kb, Bs, 32, 64,
                              (unsigned)Kd);
            tdm_wait0();
        }
        __syncthreads();

        if (kb + 256 < Kd) __builtin_prefetch(arow + kb + 256, 0, 1);

        v16h af = ld_frag(arow + kb + hi * 8, arow + kb + 16 + hi * 8);
#pragma unroll
        for (int nt = 0; nt < 4; ++nt) {
            const _Float16* bsp = Bs + (nt * 16 + lo) * 32;
            v16h bf = ld_frag(bsp + hi * 8, bsp + 16 + hi * 8);
            acc[nt] = wmma_f16(af, bf, acc[nt]);
        }
        __syncthreads();
    }

    // epilogue: lane lo = column, rows r + 8*hi
#pragma unroll
    for (int nt = 0; nt < 4; ++nt) {
        const int nn = nbase + nt * 16 + lo;
#pragma unroll
        for (int r = 0; r < 8; ++r) {
            const int row = mbase + r + 8 * hi;
            const float a = acc[nt][r];
            if (mode == 0) {
                const int sel = nn >> 10;
                const int nc  = nn & (DMODEL - 1);
                const size_t oidx = (size_t)row * DMODEL + nc;
                if (sel == 0)      Qo[oidx] = (_Float16)a;
                else if (sel == 1) Ko[oidx] = (_Float16)a;
                else if (sel == 2) Vo[oidx] = (_Float16)a;
                else {
                    const float z = a + bg[nc];
                    gateo[oidx] = 1.0f / (1.0f + __expf(-z));
                }
            } else {
                Yo[(size_t)row * N + nn] = a + bo[nn];
            }
        }
    }
}

// ---------------------------------------------------------------------------
// Kernel 4: flash attention. One wave per (batch, head, 16-query tile).
// Per 32-key step: TDM loads the K tile (32x64, row-major, no transpose
// needed) into wave-private LDS overlapping the V transpose staging;
// 4 WMMAs for S = Q K^T, online softmax (shfl_xor across 16-lane column
// groups), P via LDS round-trip, 4 WMMAs for O += P V.
// ---------------------------------------------------------------------------
__global__ __launch_bounds__(128) void flash_attn(const _Float16* __restrict__ Q,
                                                  const _Float16* __restrict__ K,
                                                  const _Float16* __restrict__ V,
                                                  const float* __restrict__ gate,
                                                  _Float16* __restrict__ Oh) {
    __shared__ __align__(16) _Float16 KtAll[4][32 * HDIM];  // Kt[key][hd]
    __shared__ __align__(16) _Float16 VtAll[4][HDIM * 32];  // Vt[hd][key]
    __shared__ __align__(16) _Float16 PtAll[4][16 * 32];    // Pt[row][key]

    const int tid  = threadIdx.x;
    const int lane = tid & 31;
    const int wave = tid >> 5;
    const int lo   = lane & 15;
    const int hi   = lane >> 4;
    _Float16* Kt = KtAll[wave];
    _Float16* Vt = VtAll[wave];
    _Float16* Pt = PtAll[wave];

    const int task  = blockIdx.x * 4 + wave;     // 4096 tasks
    const int qt    = task & 127;                // 128 q-tiles
    const int head  = (task >> 7) & (NHEAD - 1);
    const int batch = task >> 11;
    const int qbase = qt * 16;
    const size_t headoff = (size_t)head * HDIM;

    // Q fragments: hd chunks [0,32) and [32,64)
    const _Float16* qrow = Q + ((size_t)(batch * LSEQ + qbase + lo)) * DMODEL + headoff;
    v16h qf0 = ld_frag(qrow + hi * 8,      qrow + 16 + hi * 8);
    v16h qf1 = ld_frag(qrow + 32 + hi * 8, qrow + 48 + hi * 8);

    v8f o[4] = {v8f{}, v8f{}, v8f{}, v8f{}};
    float rm[8], rl[8];
#pragma unroll
    for (int r = 0; r < 8; ++r) { rm[r] = -1e30f; rl[r] = 0.0f; }
    const float scale = 0.125f;  // 1/sqrt(64)

    for (int kb = 0; kb < LSEQ; kb += 32) {
        // issue TDM for the K tile first so it overlaps the V staging below
        tdm_load_tile_f16(K + ((size_t)(batch * LSEQ + kb)) * DMODEL + headoff,
                          Kt, HDIM, 32, DMODEL);

        // stage V transposed: lane handles key kb+lane, writes column `lane`
        {
            const _Float16* vrow =
                V + ((size_t)(batch * LSEQ + kb + lane)) * DMODEL + headoff;
#pragma unroll
            for (int c = 0; c < 8; ++c) {
                v8h dq = *reinterpret_cast<const v8h*>(vrow + c * 8);
#pragma unroll
                for (int j = 0; j < 8; ++j) Vt[(c * 8 + j) * 32 + lane] = dq[j];
            }
        }

        tdm_wait0();  // K tile resident in LDS

        // S tiles for keys [kb, kb+16) and [kb+16, kb+32) from LDS Kt
        v8f s0 = v8f{}, s1 = v8f{};
        {
            const _Float16* kr0 = Kt + lo * HDIM;
            const _Float16* kr1 = Kt + (16 + lo) * HDIM;
            s0 = wmma_f16(qf0, ld_frag(kr0 + hi * 8,      kr0 + 16 + hi * 8), s0);
            s0 = wmma_f16(qf1, ld_frag(kr0 + 32 + hi * 8, kr0 + 48 + hi * 8), s0);
            s1 = wmma_f16(qf0, ld_frag(kr1 + hi * 8,      kr1 + 16 + hi * 8), s1);
            s1 = wmma_f16(qf1, ld_frag(kr1 + 32 + hi * 8, kr1 + 48 + hi * 8), s1);
        }

        // online softmax over the 32 new columns (row-wise across 16 lanes)
        float p0[8], p1[8];
#pragma unroll
        for (int r = 0; r < 8; ++r) {
            float a = s0[r] * scale, b = s1[r] * scale;
            float mx = fmaxf(a, b);
            mx = fmaxf(mx, __shfl_xor(mx, 1, 32));
            mx = fmaxf(mx, __shfl_xor(mx, 2, 32));
            mx = fmaxf(mx, __shfl_xor(mx, 4, 32));
            mx = fmaxf(mx, __shfl_xor(mx, 8, 32));
            const float mnew = fmaxf(rm[r], mx);
            const float corr = __expf(rm[r] - mnew);
            const float e0 = __expf(a - mnew);
            const float e1 = __expf(b - mnew);
            float rs = e0 + e1;
            rs += __shfl_xor(rs, 1, 32);
            rs += __shfl_xor(rs, 2, 32);
            rs += __shfl_xor(rs, 4, 32);
            rs += __shfl_xor(rs, 8, 32);
            rl[r] = rl[r] * corr + rs;
            rm[r] = mnew;
            p0[r] = e0; p1[r] = e1;
            o[0][r] *= corr; o[1][r] *= corr; o[2][r] *= corr; o[3][r] *= corr;
        }

        // P: C-layout -> LDS -> A-layout fragment (16 x 32)
#pragma unroll
        for (int r = 0; r < 8; ++r) {
            Pt[(r + 8 * hi) * 32 + lo]      = (_Float16)p0[r];
            Pt[(r + 8 * hi) * 32 + 16 + lo] = (_Float16)p1[r];
        }
        const _Float16* prow = Pt + lo * 32;
        v16h pf = ld_frag(prow + hi * 8, prow + 16 + hi * 8);

        // O += P (16x32) @ V (32x64), 16-column chunks
#pragma unroll
        for (int c = 0; c < 4; ++c) {
            const _Float16* vp = Vt + (c * 16 + lo) * 32;
            v16h vf = ld_frag(vp + hi * 8, vp + 16 + hi * 8);
            o[c] = wmma_f16(pf, vf, o[c]);
        }
    }

    // epilogue: normalize, gate, store f16
    const float* grow = gate + ((size_t)(batch * LSEQ + qbase)) * DMODEL + headoff;
    _Float16* orow    = Oh   + ((size_t)(batch * LSEQ + qbase)) * DMODEL + headoff;
#pragma unroll
    for (int c = 0; c < 4; ++c) {
        const int col = c * 16 + lo;
#pragma unroll
        for (int r = 0; r < 8; ++r) {
            const int row = r + 8 * hi;
            const float val = o[c][r] / rl[r];
            const float g = grow[(size_t)row * DMODEL + col];
            orow[(size_t)row * DMODEL + col] = (_Float16)(val * g);
        }
    }
}

// ---------------------------------------------------------------------------
// host-side launch
// ---------------------------------------------------------------------------
extern "C" void kernel_launch(void* const* d_in, const int* in_sizes, int n_in,
                              void* d_out, int out_size, void* d_ws, size_t ws_size,
                              hipStream_t stream) {
    (void)in_sizes; (void)n_in; (void)out_size; (void)ws_size;

    const float* seq   = (const float*)d_in[0];
    const float* gamma = (const float*)d_in[1];
    const float* beta  = (const float*)d_in[2];
    const float* Wq    = (const float*)d_in[3];
    const float* Wk    = (const float*)d_in[4];
    const float* Wv    = (const float*)d_in[5];
    const float* Wg    = (const float*)d_in[6];
    const float* bg    = (const float*)d_in[7];
    const float* Wo    = (const float*)d_in[8];
    const float* bo    = (const float*)d_in[9];
    float* out = (float*)d_out;

    char* ws = (char*)d_ws;
    const size_t MB = 1u << 20;
    _Float16* Xh    = (_Float16*)(ws + 0 * MB);   //  8 MB  (4096x1024 f16)
    _Float16* WallT = (_Float16*)(ws + 8 * MB);   //  8 MB  (4096x1024 f16, B^T)
    _Float16* WohT  = (_Float16*)(ws + 16 * MB);  //  2 MB  (1024x1024 f16, B^T)
    _Float16* Qh    = (_Float16*)(ws + 18 * MB);  //  8 MB
    _Float16* Kh    = (_Float16*)(ws + 26 * MB);  //  8 MB
    _Float16* Vh    = (_Float16*)(ws + 34 * MB);  //  8 MB
    float*    gateb = (float*)   (ws + 42 * MB);  // 16 MB  (4096x1024 f32)
    _Float16* AOh   = (_Float16*)(ws + 58 * MB);  //  8 MB  (gated attn out f16)

    // 1. pack / convert weights (transposed for TDM-friendly B tiles)
    {
        const int total = DMODEL * N4 + DMODEL * DMODEL;
        prep_weights<<<(total + 255) / 256, 256, 0, stream>>>(Wq, Wk, Wv, Wg, Wo,
                                                              WallT, WohT);
    }
    // 2. LayerNorm -> f16
    layernorm_f16<<<SROWS, 256, 0, stream>>>(seq, gamma, beta, Xh);

    // 3. QKVG projection GEMM (4096x1024 @ 1024x4096), split epilogue
    {
        dim3 grid(N4 / 64, SROWS / 128);
        wmma_gemm<<<grid, 256, 0, stream>>>(Xh, WallT, N4, DMODEL, 0,
                                            Qh, Kh, Vh, gateb, bg,
                                            nullptr, nullptr);
    }
    // 4. attention + gate -> AOh (f16)
    flash_attn<<<(BATCH * NHEAD * (LSEQ / 16)) / 4, 128, 0, stream>>>(
        Qh, Kh, Vh, gateb, AOh);

    // 5. output projection (4096x1024 @ 1024x1024) + bo -> f32 d_out
    {
        dim3 grid(DMODEL / 64, SROWS / 128);
        wmma_gemm<<<grid, 256, 0, stream>>>(AOh, WohT, DMODEL, DMODEL, 1,
                                            nullptr, nullptr, nullptr, nullptr,
                                            nullptr, out, bo);
    }
}
